// TransformerModel_DIS_Time_21878563406312
// MI455X (gfx1250) — compile-verified, matrix operand order checked by reading, not run on hardware
//
#include <hip/hip_runtime.h>
#include <hip/hip_bf16.h>
#include <math.h>

// ---------------- CDNA5 WMMA types ----------------
typedef __attribute__((ext_vector_type(16))) __bf16 v16bf;
typedef __attribute__((ext_vector_type(8)))  float  v8f;

union FRAG { v16bf v; uint4 q[2]; unsigned short u[16]; };

// f32 -> bf16, round-half-up (1 add); pack two with one v_perm_b32.
__device__ __forceinline__ unsigned short f2bfr(float f) {
  return (unsigned short)((__float_as_uint(f) + 0x8000u) >> 16);
}
__device__ __forceinline__ unsigned int pk2(float a, float b) {
  return __builtin_amdgcn_perm(__float_as_uint(b) + 0x8000u,
                               __float_as_uint(a) + 0x8000u, 0x07060302u);
}
__device__ __forceinline__ float bf2f(unsigned short u) {
  return __uint_as_float(((unsigned int)u) << 16);
}
// Build a bf16 fragment from 32 contiguous f32, per ISA 7.12.2 layout:
// lane half 'hi' holds K = {hi*8..+7, 16+hi*8..+7}.
__device__ __forceinline__ void frag_from_f32(FRAG& f, const float* __restrict__ p, int hi) {
  float4 a0 = *(const float4*)(p + hi * 8);
  float4 a1 = *(const float4*)(p + hi * 8 + 4);
  float4 b0 = *(const float4*)(p + 16 + hi * 8);
  float4 b1 = *(const float4*)(p + 16 + hi * 8 + 4);
  f.q[0].x = pk2(a0.x, a0.y); f.q[0].y = pk2(a0.z, a0.w);
  f.q[0].z = pk2(a1.x, a1.y); f.q[0].w = pk2(a1.z, a1.w);
  f.q[1].x = pk2(b0.x, b0.y); f.q[1].y = pk2(b0.z, b0.w);
  f.q[1].z = pk2(b1.x, b1.y); f.q[1].w = pk2(b1.z, b1.w);
}

static constexpr int Sd = 256, Bd = 4, INd = 64, Hd = 4, HIDd = 256, HDd = 64;
static constexpr float NEGF = -4294967295.0f;

// ---------------- one-time f32 -> bf16 conversion WITH transpose ----------------
// Y[n*K + k] = bf16(X[k*N + n]); grid = N blocks.
__global__ __launch_bounds__(256)
void k_cvtT(const float* __restrict__ X, unsigned short* __restrict__ Y,
            int K, int N) {
  int n = blockIdx.x;
  for (int k = threadIdx.x; k < K; k += blockDim.x)
    Y[n * K + k] = f2bfr(X[k * N + n]);
}

// ---------------- prep: transpose [S,B,*] -> [B*S,*] ----------------
__global__ __launch_bounds__(64)
void k_prep(const float* __restrict__ src_inp, const float* __restrict__ timev,
            float* __restrict__ xT, float* __restrict__ tfl) {
  int bs = blockIdx.x;                 // b*S + s
  int b = bs >> 8, s = bs & 255;
  int c = threadIdx.x;                 // 0..63
  xT[bs * INd + c] = src_inp[(s * Bd + b) * INd + c];
  if (c == 0) tfl[bs] = timev[s * Bd + b];
}

// ---------------- cumsum of relu(t) per batch ----------------
__global__ void k_pos(const float* __restrict__ tfl, float* __restrict__ pos) {
  int b = threadIdx.x;
  if (b >= Bd) return;
  float acc = 0.f;
  for (int s = 0; s < Sd; ++s) {
    float t = tfl[b * Sd + s];
    acc += (t > 0.f ? t : 0.f);
    pos[b * Sd + s] = acc;
  }
}

// ---------------- LayerNorm (optional residual add of X2) ----------------
template <bool ADD>
__global__ __launch_bounds__(256)
void k_ln(const float* __restrict__ X, const float* __restrict__ X2,
          const float* __restrict__ sc, const float* __restrict__ bi,
          float* __restrict__ Y) {
  __shared__ float red[256];
  int row = blockIdx.x, d = threadIdx.x;
  float x = X[row * HIDd + d];
  if (ADD) x += X2[row * HIDd + d];
  red[d] = x; __syncthreads();
  for (int off = 128; off > 0; off >>= 1) { if (d < off) red[d] += red[d + off]; __syncthreads(); }
  float m = red[0] * (1.f / HIDd);
  __syncthreads();
  float c = x - m;
  red[d] = c * c; __syncthreads();
  for (int off = 128; off > 0; off >>= 1) { if (d < off) red[d] += red[d + off]; __syncthreads(); }
  float v = red[0] * (1.f / HIDd);
  Y[row * HIDd + d] = c * rsqrtf(v + 1e-8f) * sc[d] + bi[d];
}

// ---------------- LDS-free WMMA bf16 GEMM ----------------
// C = act(A@W + bias [+tvec*tw+tb] [+resid]); A:[M,K] f32, WbT:[N,K] bf16 (pre-transposed).
// OM: 0 = f32 out [M,N]; 1 = bf16 out [M,N]; 2 = bf16 out transposed [B,H,HD,S].
// Block tile 64x64, 128 thr = 4 waves; fragments straight from global (L2-resident).
template <int OM, bool HAS_T, bool RELU, bool HAS_RES>
__global__ __launch_bounds__(128)
void k_gemm(const float* __restrict__ A, const unsigned short* __restrict__ WbT,
            const float* __restrict__ bias, void* __restrict__ Cv,
            int K, int N,
            const float* __restrict__ tvec, const float* __restrict__ tw,
            const float* __restrict__ tb,
            const float* __restrict__ resid) {
  int mBase = blockIdx.x * 64;
  int nBase = blockIdx.y * 64;
  int tid = threadIdx.x;
  int wv = tid >> 5, lane = tid & 31, l15 = lane & 15, hi = lane >> 4;
  int arow = mBase + wv * 16 + l15;
  v8f acc[4] = {};
  for (int kc = 0; kc < K; kc += 32) {
    FRAG af;
    frag_from_f32(af, A + arow * K + kc, hi);
    #pragma unroll
    for (int nt = 0; nt < 4; ++nt) {
      const unsigned short* wp = WbT + (nBase + nt * 16 + l15) * K + kc;
      FRAG bf;
      bf.q[0] = *(const uint4*)(wp + hi * 8);
      bf.q[1] = *(const uint4*)(wp + 16 + hi * 8);
      acc[nt] = __builtin_amdgcn_wmma_f32_16x16x32_bf16(
          false, af.v, false, bf.v, (short)0, acc[nt], false, false);
    }
  }
  float tv8[8];
  if (HAS_T) {
    #pragma unroll
    for (int i = 0; i < 8; ++i) tv8[i] = tvec[mBase + wv * 16 + hi * 8 + i];
  }
  #pragma unroll
  for (int nt = 0; nt < 4; ++nt) {
    int n = nBase + nt * 16 + l15;
    float bv = bias[n];
    float twv = HAS_T ? tw[n] : 0.f;
    float tbv = HAS_T ? tb[n] : 0.f;
    #pragma unroll
    for (int i = 0; i < 8; ++i) {
      int m = mBase + wv * 16 + hi * 8 + i;
      float v = acc[nt][i] + bv;
      if (HAS_T) v += tv8[i] * twv + tbv;
      if (RELU) v = v > 0.f ? v : 0.f;
      if (OM == 0) {
        float o = v;
        if (HAS_RES) o += resid[m * N + n];
        ((float*)Cv)[m * N + n] = o;
      } else if (OM == 1) {
        ((unsigned short*)Cv)[m * N + n] = f2bfr(v);
      } else {  // OM == 2: [B,H,HD,S] bf16 (for attention V' B-fragments)
        int idx = (((m >> 8) * Hd + (n >> 6)) * HDd + (n & 63)) * Sd + (m & 255);
        ((unsigned short*)Cv)[idx] = f2bfr(v);
      }
    }
  }
}

// ---------------- per-(row,head) dot of bf16 Q with tiK_w / tiK_b ----------------
__global__ __launch_bounds__(256)
void k_qdots(const unsigned short* __restrict__ Q, const float* __restrict__ tw,
             const float* __restrict__ tb, float* __restrict__ qdw,
             float* __restrict__ qdb) {
  int idx = blockIdx.x * blockDim.x + threadIdx.x;     // row*H + h
  if (idx >= Bd * Sd * Hd) return;
  int row = idx >> 2, h = idx & 3;
  const unsigned short* q = Q + row * HIDd + h * HDd;
  float sw = 0.f, sb = 0.f;
  for (int d = 0; d < HDd; ++d) {
    float qv = bf2f(q[d]);
    sw += qv * tw[h * HDd + d];
    sb += qv * tb[h * HDd + d];
  }
  qdw[idx] = sw; qdb[idx] = sb;
}

// ---------------- fused attention: scores -> softmax -> A@V' ----------------
// Q/K' bf16 [BS,HID]; VT bf16 [B,H,HD,S]. grid: B*H*8 blocks (32 q-rows), 128 thr.
__global__ __launch_bounds__(128)
void k_attn(const unsigned short* __restrict__ Q, const unsigned short* __restrict__ Kp,
            const unsigned short* __restrict__ VT, const float* __restrict__ pos,
            const float* __restrict__ qdw, const float* __restrict__ qdb,
            const float* __restrict__ tivw, const float* __restrict__ tivb,
            float* __restrict__ out) {
  __shared__ float sc[32 * 256];                        // f32 scores (32KB)
  __shared__ __align__(16) unsigned short sa[32 * 256]; // bf16 softmaxed A (16KB)
  __shared__ float stat[64];
  __shared__ float red2[128];
  int blk = blockIdx.x;
  int qb = blk & 7, h = (blk >> 3) & 3, b = blk >> 5;
  int qBase = qb * 32, bS = b * Sd;
  int tid = threadIdx.x, wv = tid >> 5, lane = tid & 31, l15 = lane & 15, hi = lane >> 4;
  const float scale = 0.125f;                           // 1/sqrt(HD)

  // ---- phase 1: scores = (Q@K'^T + dist*qdw + qdb)*scale, causal-masked ----
  int qs = wv >> 1, kh = wv & 1;                        // q-strip of 16, k-half of 128
  FRAG a0, a1;
  {
    const unsigned short* qp = Q + (bS + qBase + qs * 16 + l15) * HIDd + h * HDd;
    a0.q[0] = *(const uint4*)(qp + hi * 8);
    a0.q[1] = *(const uint4*)(qp + 16 + hi * 8);
    a1.q[0] = *(const uint4*)(qp + 32 + hi * 8);
    a1.q[1] = *(const uint4*)(qp + 48 + hi * 8);
  }
  float posq8[8], qdw8[8], qdb8[8];
  #pragma unroll
  for (int i = 0; i < 8; ++i) {
    int q = qBase + qs * 16 + hi * 8 + i;
    posq8[i] = pos[bS + q];
    qdw8[i] = qdw[(bS + q) * Hd + h];
    qdb8[i] = qdb[(bS + q) * Hd + h];
  }
  for (int nt = 0; nt < 8; ++nt) {
    int colBase = kh * 128 + nt * 16;
    int key = colBase + l15;
    const unsigned short* kp = Kp + (bS + key) * HIDd + h * HDd;
    FRAG b0, b1;
    b0.q[0] = *(const uint4*)(kp + hi * 8);
    b0.q[1] = *(const uint4*)(kp + 16 + hi * 8);
    b1.q[0] = *(const uint4*)(kp + 32 + hi * 8);
    b1.q[1] = *(const uint4*)(kp + 48 + hi * 8);
    v8f acc = {};
    acc = __builtin_amdgcn_wmma_f32_16x16x32_bf16(false, a0.v, false, b0.v, (short)0, acc, false, false);
    acc = __builtin_amdgcn_wmma_f32_16x16x32_bf16(false, a1.v, false, b1.v, (short)0, acc, false, false);
    float pk = pos[bS + key];
    #pragma unroll
    for (int i = 0; i < 8; ++i) {
      int rl = qs * 16 + hi * 8 + i;                    // local q row (0..31)
      int q = qBase + rl;
      float dist = fabsf(posq8[i] - pk);
      float v = (acc[i] + dist * qdw8[i] + qdb8[i]) * scale;
      if (key > q) v = NEGF;                            // mask above diagonal
      sc[rl * 256 + key] = v;
    }
  }
  __syncthreads();

  // ---- phase 2: softmax (4 threads/row) + Sum(A*dist); A -> sa (bf16) ----
  {
    int rl = tid >> 2, seg = tid & 3;
    float* row = sc + rl * 256 + seg * 64;
    float m = -3.4e38f;
    for (int k = 0; k < 64; ++k) m = fmaxf(m, row[k]);
    red2[tid] = m; __syncthreads();
    m = fmaxf(fmaxf(red2[rl * 4], red2[rl * 4 + 1]),
              fmaxf(red2[rl * 4 + 2], red2[rl * 4 + 3]));
    float pq = pos[bS + qBase + rl];
    const float* pk = pos + bS + seg * 64;
    float sum = 0.f, sumd = 0.f;
    for (int k = 0; k < 64; ++k) {
      float e = __expf(row[k] - m);
      row[k] = e;
      sum += e;
      sumd += e * fabsf(pq - pk[k]);
    }
    __syncthreads();
    red2[tid] = sum; __syncthreads();
    sum = red2[rl * 4] + red2[rl * 4 + 1] + red2[rl * 4 + 2] + red2[rl * 4 + 3];
    __syncthreads();
    red2[tid] = sumd; __syncthreads();
    sumd = red2[rl * 4] + red2[rl * 4 + 1] + red2[rl * 4 + 2] + red2[rl * 4 + 3];
    float inv = 1.f / sum;
    unsigned short* dst = sa + rl * 256 + seg * 64;
    for (int k = 0; k < 64; k += 2)
      *(unsigned int*)&dst[k] = pk2(row[k] * inv, row[k + 1] * inv);
    if (seg == 0) { stat[rl] = sumd * inv; stat[32 + rl] = 1.0f; }
  }
  __syncthreads();

  // ---- phase 3: out = A @ V' + stat_d*tiV_w + stat_s*tiV_b ----
  // B-fragments come straight from global VT[b,h,d,s] (pre-transposed by V' GEMM).
  int dn = (wv & 1) * 32;
  const unsigned short* vtp = VT + (b * Hd + h) * HDd * Sd;
  v8f acc2[2] = {};
  for (int kc = 0; kc < Sd; kc += 32) {
    FRAG af;
    int arow = qs * 16 + l15;
    af.q[0] = *(const uint4*)&sa[arow * 256 + kc + hi * 8];
    af.q[1] = *(const uint4*)&sa[arow * 256 + kc + 16 + hi * 8];
    #pragma unroll
    for (int nt = 0; nt < 2; ++nt) {
      int dcol = dn + nt * 16 + l15;
      const unsigned short* vp = vtp + dcol * Sd + kc;
      FRAG bf;
      bf.q[0] = *(const uint4*)(vp + hi * 8);
      bf.q[1] = *(const uint4*)(vp + 16 + hi * 8);
      acc2[nt] = __builtin_amdgcn_wmma_f32_16x16x32_bf16(
          false, af.v, false, bf.v, (short)0, acc2[nt], false, false);
    }
  }
  #pragma unroll
  for (int nt = 0; nt < 2; ++nt) {
    int dcol = dn + nt * 16 + l15;
    float wv2 = tivw[h * HDd + dcol], bv2 = tivb[h * HDd + dcol];
    #pragma unroll
    for (int i = 0; i < 8; ++i) {
      int rl = qs * 16 + hi * 8 + i;
      int q = qBase + rl;
      out[(bS + q) * HIDd + h * HDd + dcol] = acc2[nt][i] + stat[rl] * wv2 + stat[32 + rl] * bv2;
    }
  }
}

// ---------------- final: sigmoid(LN'd row @ dec_w + dec_b), mean over S ----------------
__global__ __launch_bounds__(256)
void k_decode(const float* __restrict__ X, const float* __restrict__ dw,
              const float* __restrict__ db, float* __restrict__ out) {
  __shared__ float red[256];
  int b = blockIdx.x, s = threadIdx.x;
  const float* x = X + (b * Sd + s) * HIDd;
  float dot = db[0];
  for (int d = 0; d < HIDd; ++d) dot += x[d] * dw[d];
  float sg = 1.f / (1.f + __expf(-dot));
  red[s] = sg; __syncthreads();
  for (int off = 128; off > 0; off >>= 1) { if (s < off) red[s] += red[s + off]; __syncthreads(); }
  if (s == 0) out[b] = red[0] * (1.f / Sd);
}

// ---------------- host orchestration ----------------
extern "C" void kernel_launch(void* const* d_in, const int* in_sizes, int n_in,
                              void* d_out, int out_size, void* d_ws, size_t ws_size,
                              hipStream_t stream) {
  const float* src_inp = (const float*)d_in[0];
  const float* timev   = (const float*)d_in[1];
  const float* seq_w   = (const float*)d_in[3];
  const float* seq_b   = (const float*)d_in[4];
  const float* tK_w    = (const float*)d_in[5];
  const float* tK_b    = (const float*)d_in[6];
  const float* tV_w    = (const float*)d_in[7];
  const float* tV_b    = (const float*)d_in[8];
  const float* tiK_w   = (const float*)d_in[9];
  const float* tiK_b   = (const float*)d_in[10];
  const float* tiV_w   = (const float*)d_in[11];
  const float* tiV_b   = (const float*)d_in[12];
  const float* ln1_s   = (const float*)d_in[13];
  const float* ln1_b   = (const float*)d_in[14];
  const float* qw      = (const float*)d_in[15];
  const float* qb      = (const float*)d_in[16];
  const float* kw      = (const float*)d_in[17];
  const float* kb      = (const float*)d_in[18];
  const float* vw      = (const float*)d_in[19];
  const float* vb      = (const float*)d_in[20];
  const float* ln2_s   = (const float*)d_in[21];
  const float* ln2_b   = (const float*)d_in[22];
  const float* c1_w    = (const float*)d_in[23];
  const float* c1_b    = (const float*)d_in[24];
  const float* c2_w    = (const float*)d_in[25];
  const float* c2_b    = (const float*)d_in[26];
  const float* lnf_s   = (const float*)d_in[27];
  const float* lnf_b   = (const float*)d_in[28];
  const float* dec_w   = (const float*)d_in[29];
  const float* dec_b   = (const float*)d_in[30];

  const int BS = Bd * Sd;                       // 1024
  const int MAT = BS * HIDd;                    // 262144
  const int WSZ = HIDd * HIDd;                  // 65536 per layer
  char* p = (char*)d_ws;
  auto allocF = [&](size_t n) { float* r = (float*)p; p += n * 4; return r; };
  auto allocH = [&](size_t n) { unsigned short* r = (unsigned short*)p; p += n * 2; return r; };

  float* xT   = allocF(BS * INd);
  float* tfl  = allocF(BS);
  float* pos  = allocF(BS);
  float* src  = allocF(MAT);
  float* Qn   = allocF(MAT);
  float* att  = allocF(MAT);
  float* sr2  = allocF(MAT);
  float* hb   = allocF(MAT);
  float* qdw  = allocF(BS * Hd);
  float* qdb  = allocF(BS * Hd);
  unsigned short* seqwT = allocH(INd * HIDd);   // [N=256][K=64]
  unsigned short* qwT   = allocH(2 * WSZ);      // per layer [N][K]
  unsigned short* kwT   = allocH(2 * WSZ);
  unsigned short* vwT   = allocH(2 * WSZ);
  unsigned short* c1T   = allocH(2 * WSZ);
  unsigned short* c2T   = allocH(2 * WSZ);
  unsigned short* QB    = allocH(MAT);          // [BS,HID] bf16
  unsigned short* KB    = allocH(MAT);          // [BS,HID] bf16
  unsigned short* VT    = allocH(MAT);          // [B,H,HD,S] bf16
  if (ws_size < (size_t)(p - (char*)d_ws)) return;

  dim3 gGemm(BS / 64, HIDd / 64);               // 16 x 4

  // one-time weight conversion + transpose (f32 [K,N] -> bf16 [N,K])
  k_cvtT<<<HIDd, 256, 0, stream>>>(seq_w, seqwT, INd, HIDd);
  for (int i = 0; i < 2; ++i) {
    k_cvtT<<<HIDd, 256, 0, stream>>>(qw + i * WSZ, qwT + i * WSZ, HIDd, HIDd);
    k_cvtT<<<HIDd, 256, 0, stream>>>(kw + i * WSZ, kwT + i * WSZ, HIDd, HIDd);
    k_cvtT<<<HIDd, 256, 0, stream>>>(vw + i * WSZ, vwT + i * WSZ, HIDd, HIDd);
    k_cvtT<<<HIDd, 256, 0, stream>>>(c1_w + i * WSZ, c1T + i * WSZ, HIDd, HIDd);
    k_cvtT<<<HIDd, 256, 0, stream>>>(c2_w + i * WSZ, c2T + i * WSZ, HIDd, HIDd);
  }

  k_prep<<<BS, 64, 0, stream>>>(src_inp, timev, xT, tfl);
  k_pos<<<1, Bd, 0, stream>>>(tfl, pos);
  k_gemm<0, false, false, false><<<gGemm, 128, 0, stream>>>(
      xT, seqwT, seq_b, src, INd, HIDd, nullptr, nullptr, nullptr, nullptr);

  for (int i = 0; i < 2; ++i) {
    const int VO = i * HIDd;
    k_ln<false><<<BS, 256, 0, stream>>>(src, nullptr, ln1_s + VO, ln1_b + VO, Qn);
    k_gemm<1, false, false, false><<<gGemm, 128, 0, stream>>>(
        Qn, qwT + i * WSZ, qb + VO, QB, HIDd, HIDd, nullptr, nullptr, nullptr, nullptr);
    k_gemm<1, true, false, false><<<gGemm, 128, 0, stream>>>(    // K' = K+absK
        src, kwT + i * WSZ, kb + VO, KB, HIDd, HIDd, tfl, tK_w, tK_b, nullptr);
    k_gemm<2, true, false, false><<<gGemm, 128, 0, stream>>>(    // V' = V+absV (transposed out)
        src, vwT + i * WSZ, vb + VO, VT, HIDd, HIDd, tfl, tV_w, tV_b, nullptr);
    k_qdots<<<(BS * Hd + 255) / 256, 256, 0, stream>>>(QB, tiK_w, tiK_b, qdw, qdb);
    k_attn<<<Bd * Hd * 8, 128, 0, stream>>>(QB, KB, VT, pos, qdw, qdb,
                                            tiV_w, tiV_b, att);
    k_ln<true><<<BS, 256, 0, stream>>>(Qn, att, ln2_s + VO, ln2_b + VO, sr2);
    k_gemm<0, false, true, false><<<gGemm, 128, 0, stream>>>(    // relu FFN1
        sr2, c1T + i * WSZ, c1_b + VO, hb, HIDd, HIDd, nullptr, nullptr, nullptr, nullptr);
    k_gemm<0, false, false, true><<<gGemm, 128, 0, stream>>>(    // FFN2 + resid
        hb, c2T + i * WSZ, c2_b + VO, src, HIDd, HIDd, nullptr, nullptr, nullptr, sr2);
  }

  k_ln<false><<<BS, 256, 0, stream>>>(src, nullptr, lnf_s, lnf_b, Qn);
  k_decode<<<Bd, 256, 0, stream>>>(Qn, dec_w, dec_b, (float*)d_out);
}
